// ContrastiveLoss_71665824301682
// MI455X (gfx1250) — matrix operand order repaired and checked by reading, not compiled, not run on hardware
//
#include <hip/hip_runtime.h>
#include <hip/hip_bf16.h>
#include <cstdint>
#include <cstddef>

// ---------------------------------------------------------------------------
// Problem constants (match reference: B=32, L=1024, D=512, T=0.1)
// ---------------------------------------------------------------------------
#define B_   32
#define L_   1024
#define D_   512
#define INV_T 10.0f
// Fixed LSE shift: logits = 10*cos <= ~10.1 (unit vectors, bf16 rounding),
// so exp(logit - SHIFT) <= e^0.04 -- no overflow, no running max needed.
#define SHIFT_ 10.2f

typedef __bf16 v16bf __attribute__((ext_vector_type(16)));
typedef __bf16 v8bf  __attribute__((ext_vector_type(8)));
typedef __bf16 v4bf  __attribute__((ext_vector_type(4)));
typedef float  v8f   __attribute__((ext_vector_type(8)));

union V16 { v16bf v; v8bf h[2]; };

// ---------------------------------------------------------------------------
// Kernel 1: L2-normalize rows, emit bf16 features + inverse norms.
// One 128-thread block per (b,l) row; each thread handles a float4.
// ---------------------------------------------------------------------------
__global__ __launch_bounds__(128) void nrm_kernel(const float* __restrict__ in,
                                                  __bf16* __restrict__ fbf,
                                                  float* __restrict__ invn) {
  const int row = blockIdx.x;                 // 0 .. B*L-1
  const int tid = threadIdx.x;                // 0 .. 127
  const float4 v = ((const float4*)(in + (size_t)row * D_))[tid];
  float ss = v.x * v.x + v.y * v.y + v.z * v.z + v.w * v.w;
  for (int off = 16; off > 0; off >>= 1) ss += __shfl_xor(ss, off, 32);
  __shared__ float red[4];
  const int wave = tid >> 5, lane = tid & 31;
  if (lane == 0) red[wave] = ss;
  __syncthreads();
  ss = red[0] + red[1] + red[2] + red[3];
  const float inv = 1.0f / fmaxf(sqrtf(ss), 1e-12f);
  v4bf o;
  o[0] = (__bf16)(v.x * inv);
  o[1] = (__bf16)(v.y * inv);
  o[2] = (__bf16)(v.z * inv);
  o[3] = (__bf16)(v.w * inv);
  *(v4bf*)(fbf + (size_t)row * D_ + tid * 4) = o;
  if (tid == 0) invn[row] = inv;
}

// ---------------------------------------------------------------------------
// Kernel 2: dataset-id metadata: first positive index + validity per anchor.
// One block per batch; ids cached in LDS.
// ---------------------------------------------------------------------------
__global__ __launch_bounds__(256) void meta_kernel(const int* __restrict__ ids,
                                                   int* __restrict__ first_pos,
                                                   int* __restrict__ valid) {
  const int b = blockIdx.x;
  const int tid = threadIdx.x;
  __shared__ int sid[L_];
  for (int i = tid; i < L_; i += 256) sid[i] = ids[b * L_ + i];
  __syncthreads();
  for (int j = 0; j < L_ / 256; ++j) {
    const int l = tid + 256 * j;
    const int id = sid[l];
    int cnt = 0, first = L_;
    for (int mm = 0; mm < L_; ++mm) {
      if (sid[mm] == id) { ++cnt; if (first == L_) first = mm; }
    }
    const int g = b * L_ + l;
    first_pos[g] = first;
    valid[g] = (cnt > 1 && cnt < L_) ? 1 : 0;   // pos_count>1 && neg_count>0
  }
}

// ---------------------------------------------------------------------------
// Kernel 3: target logit = cos(f_l, f_firstpos) / T in full f32.
// One thread per anchor.
// ---------------------------------------------------------------------------
__global__ __launch_bounds__(256) void tgt_kernel(const float* __restrict__ in,
                                                  const float* __restrict__ invn,
                                                  const int* __restrict__ first_pos,
                                                  float* __restrict__ target) {
  const int g = blockIdx.x * 256 + threadIdx.x;   // 0 .. B*L-1
  const int b = g >> 10;
  const int l = g & (L_ - 1);
  const int fp = first_pos[g];
  const float4* x = (const float4*)(in + ((size_t)b * L_ + l) * D_);
  const float4* y = (const float4*)(in + ((size_t)b * L_ + fp) * D_);
  float acc = 0.0f;
#pragma unroll 8
  for (int i = 0; i < D_ / 4; ++i) {
    const float4 a = x[i], c = y[i];
    acc += a.x * c.x + a.y * c.y + a.z * c.z + a.w * c.w;
  }
  target[g] = acc * invn[g] * invn[(b << 10) + fp] * INV_T;
}

// ---------------------------------------------------------------------------
// Kernel 4: WMMA Gram matrix + fixed-shift streaming log-sum-exp.
// Grid = B*4 blocks x 256 threads (8 waves). Each wave owns a 32-row strip:
//   - A fragments for TWO 16-row tiles, full K=512, resident (256 VGPRs)
//   - depth-2 software-pipelined B fragments: load (k+2) while two chained
//     v_wmma_f32_16x16x32_bf16 consume k  (512 B of B data per WMMA)
//   - bounded logits => fixed-shift LSE: s += exp(10*c - 10.2); no running
//     max, no rescale FMAs, plain shuffle-sum epilogue.
// A fragment: lane<16 holds M=lane, K={0-7,16-23}; lane>=16 M=lane-16,
//             K={8-15,24-31}  -> two 16B contiguous loads.
// B fragment: lane holds column N=lane&15 (token row n0+N), K same as A:
//             lane<16 K={0-7,16-23}, lane>=16 K={8-15,24-31} -> two 16B loads.
// ---------------------------------------------------------------------------
__global__ __launch_bounds__(256) void lse_kernel(const __bf16* __restrict__ F,
                                                  const float* __restrict__ target,
                                                  const int* __restrict__ valid,
                                                  float* __restrict__ per_anchor) {
  const int b     = blockIdx.x >> 2;          // 4 blocks per batch
  const int strip = blockIdx.x & 3;           // 256-row strip per block
  const int wave  = threadIdx.x >> 5;
  const int lane  = threadIdx.x & 31;
  const int row_base = strip * 256 + wave * 32;   // this wave: 32 rows
  const __bf16* Fb = F + (size_t)b * L_ * D_;

  // Preload A fragments for both 16-row tiles, all 16 k-steps (K=512).
  V16 a0[16], a1[16];
  {
    const int akoff = (lane & 16) ? 8 : 0;
    const __bf16* ar0 = Fb + (size_t)(row_base + (lane & 15)) * D_ + akoff;
    const __bf16* ar1 = ar0 + (size_t)16 * D_;
#pragma unroll
    for (int k = 0; k < 16; ++k) {
      a0[k].h[0] = *(const v8bf*)(ar0 + k * 32);
      a0[k].h[1] = *(const v8bf*)(ar0 + k * 32 + 16);
      a1[k].h[0] = *(const v8bf*)(ar1 + k * 32);
      a1[k].h[1] = *(const v8bf*)(ar1 + k * 32 + 16);
    }
  }

  float s0[8], s1[8];
#pragma unroll
  for (int r = 0; r < 8; ++r) { s0[r] = 0.0f; s1[r] = 0.0f; }

  const __bf16* bbase = Fb + (size_t)(lane & 15) * D_ + ((lane & 16) ? 8 : 0);

  // Depth-2 pipeline prologue: fragments (t=0,k=0) and (t=0,k=1).
  V16 cur, n1;
  cur.h[0] = *(const v8bf*)(bbase);
  cur.h[1] = *(const v8bf*)(bbase + 16);
  n1.h[0]  = *(const v8bf*)(bbase + 32);
  n1.h[1]  = *(const v8bf*)(bbase + 48);

  for (int t = 0; t < L_ / 16; ++t) {           // 64 column tiles
    const __bf16* bp  = bbase + (size_t)t * 16 * D_;
    const int tn = (t < L_ / 16 - 1) ? t + 1 : t;     // clamp (reload, harmless)
    const __bf16* bpn = bbase + (size_t)tn * 16 * D_;
    v8f c0 = {0.f, 0.f, 0.f, 0.f, 0.f, 0.f, 0.f, 0.f};
    v8f c1 = {0.f, 0.f, 0.f, 0.f, 0.f, 0.f, 0.f, 0.f};
#pragma unroll
    for (int k = 0; k < 16; ++k) {              // K = 512 in steps of 32
      // Issue load for fragment (k+2) before consuming fragment k.
      const __bf16* np = (k < 14) ? (bp + (k + 2) * 32)
                                  : (bpn + (k - 14) * 32);
      V16 n2;
      n2.h[0] = *(const v8bf*)(np);
      n2.h[1] = *(const v8bf*)(np + 16);
      c0 = __builtin_amdgcn_wmma_f32_16x16x32_bf16(false, a0[k].v, false, cur.v,
                                                   (short)0, c0, false, false);
      c1 = __builtin_amdgcn_wmma_f32_16x16x32_bf16(false, a1[k].v, false, cur.v,
                                                   (short)0, c1, false, false);
      cur = n1;
      n1 = n2;
    }
    // Fixed-shift LSE accumulate; c0[r]/c1[r] hold rows (row_base + {0,16} +
    // r + 8*(lane>=16)), column n0 + (lane&15).
#pragma unroll
    for (int r = 0; r < 8; ++r) {
      s0[r] += __expf(fmaf(c0[r], INV_T, -SHIFT_));
      s1[r] += __expf(fmaf(c1[r], INV_T, -SHIFT_));
    }
  }

  // Sum s across the 16 lanes of each half (xor masks stay in-half).
#pragma unroll
  for (int r = 0; r < 8; ++r) {
    float sr = s0[r], tr = s1[r];
#pragma unroll
    for (int off = 1; off < 16; off <<= 1) {
      sr += __shfl_xor(sr, off, 32);
      tr += __shfl_xor(tr, off, 32);
    }
    s0[r] = sr; s1[r] = tr;
  }

  if ((lane & 15) == 0) {
    const int half = lane >> 4;
#pragma unroll
    for (int r = 0; r < 8; ++r) {
      {
        const int row = row_base + r + 8 * half;
        const int g = b * L_ + row;
        const float lse = SHIFT_ + __logf(s0[r]);
        per_anchor[g] = valid[g] ? (lse - target[g]) : 0.0f;
      }
      {
        const int row = row_base + 16 + r + 8 * half;
        const int g = b * L_ + row;
        const float lse = SHIFT_ + __logf(s1[r]);
        per_anchor[g] = valid[g] ? (lse - target[g]) : 0.0f;
      }
    }
  }
}

// ---------------------------------------------------------------------------
// Kernel 5: deterministic final reduction -> scalar loss.
// ---------------------------------------------------------------------------
__global__ __launch_bounds__(256) void fin_kernel(const float* __restrict__ per_anchor,
                                                  const int* __restrict__ valid,
                                                  float* __restrict__ out) {
  const int tid = threadIdx.x;
  float lsum = 0.0f;
  int cnt = 0;
  for (int i = tid; i < B_ * L_; i += 256) {
    lsum += per_anchor[i];
    cnt += valid[i];
  }
  for (int off = 16; off > 0; off >>= 1) {
    lsum += __shfl_xor(lsum, off, 32);
    cnt  += __shfl_xor(cnt, off, 32);
  }
  __shared__ float rs[8];
  __shared__ int   rc[8];
  const int wave = tid >> 5, lane = tid & 31;
  if (lane == 0) { rs[wave] = lsum; rc[wave] = cnt; }
  __syncthreads();
  if (tid == 0) {
    float ts = 0.0f; int tc = 0;
    for (int w = 0; w < 8; ++w) { ts += rs[w]; tc += rc[w]; }
    out[0] = ts / (float)(tc > 0 ? tc : 1);
  }
}

// ---------------------------------------------------------------------------
// Host entry
// ---------------------------------------------------------------------------
extern "C" void kernel_launch(void* const* d_in, const int* in_sizes, int n_in,
                              void* d_out, int out_size, void* d_ws, size_t ws_size,
                              hipStream_t stream) {
  (void)in_sizes; (void)n_in; (void)out_size; (void)ws_size;
  const float* features = (const float*)d_in[0];
  const int*   ids      = (const int*)d_in[1];
  float*       out      = (float*)d_out;

  // Workspace carve-up (~34.2 MB total)
  char* ws = (char*)d_ws;
  size_t off = 0;
  __bf16* fbf       = (__bf16*)(ws + off); off += (size_t)B_ * L_ * D_ * sizeof(__bf16);
  float*  invn      = (float*)(ws + off);  off += (size_t)B_ * L_ * sizeof(float);
  int*    first_pos = (int*)(ws + off);    off += (size_t)B_ * L_ * sizeof(int);
  int*    valid     = (int*)(ws + off);    off += (size_t)B_ * L_ * sizeof(int);
  float*  target    = (float*)(ws + off);  off += (size_t)B_ * L_ * sizeof(float);
  float*  per_anch  = (float*)(ws + off);  off += (size_t)B_ * L_ * sizeof(float);

  nrm_kernel<<<B_ * L_, 128, 0, stream>>>(features, fbf, invn);
  meta_kernel<<<B_, 256, 0, stream>>>(ids, first_pos, valid);
  tgt_kernel<<<(B_ * L_) / 256, 256, 0, stream>>>(features, invn, first_pos, target);
  lse_kernel<<<B_ * 4, 256, 0, stream>>>(fbf, target, valid, per_anch);
  fin_kernel<<<1, 256, 0, stream>>>(per_anch, valid, out);
}
